// DOLPHIN_ENCODER_37400575214087
// MI455X (gfx1250) — compile-verified, matrix operand order checked by reading, not run on hardware
//
#include <hip/hip_runtime.h>
#include <hip/hip_bf16.h>

// ---------------------------------------------------------------------------
// Conformer encoder for MI455X (gfx1250, wave32, WMMA bf16 -> f32 accumulate)
// ---------------------------------------------------------------------------

#define L_   6
#define T_   1024
#define D_   512
#define H_   8
#define DK_  64
#define DFF_ 2048
#define DCG_ 2048
#define CGH_ 1024
#define KS_  31
#define NDE_ 6
#define T2_  (2 * T_ - 1)   // 2047

typedef __attribute__((ext_vector_type(16))) __bf16 v16bf;
typedef __attribute__((ext_vector_type(8)))  float  v8f;

__device__ __forceinline__ unsigned short f32_to_bf16(float f) {
  unsigned int u = __float_as_uint(f);
  unsigned int r = u + 0x7FFFu + ((u >> 16) & 1u);   // round-to-nearest-even
  return (unsigned short)(r >> 16);
}
__device__ __forceinline__ unsigned int pack_bf16x2(float a, float b) {
  return (unsigned int)f32_to_bf16(a) | ((unsigned int)f32_to_bf16(b) << 16);
}

// ---------------------------------------------------------------------------
// General WMMA GEMM:  C[m*ldcm + n*ldcn] = resid + scale * act(A·B + bias)
//   - A: fp32 [M,K] row-major (lda), batched by blockIdx.z (sAz)
//   - B: fp32; bT=0 -> [K,N] (ldb row stride), bT=1 -> [N,K] (B^T product)
//   - bias: per-column (may be null), batched (sBiz)
//   - resid: fp32 [M,N] (ldr), may be null
//   - act: 0=none, 1=SiLU, 2=GELU(exact)
// Block: 128 threads (4 waves). Tile 128x64, K-step 32. Each wave: 32 rows
// (2 A-frags) x 64 cols (4 B-frags) = 8 WMMAs per K-step, B-frag reuse x2.
// waves_per_eu(4) caps VGPRs <= 256/wave: no s_set_vgpr_msb churn, 4+ waves
// per SIMD32 for latency hiding of the ds_load -> wmma chain.
// Requirements: K % 32 == 0; A/B rows 16B-aligned (all call sites satisfy).
// ---------------------------------------------------------------------------
#define BM_ 128
#define BN_ 64
#define LDSA 34
#define LDSB 34

__attribute__((amdgpu_waves_per_eu(4)))
__launch_bounds__(128)
__global__ void gemm_wmma(
    const float* __restrict__ A, long lda, long sAz,
    const float* __restrict__ B, long ldb, long sBz,
    const float* __restrict__ bias, long sBiz,
    const float* __restrict__ resid, long ldr, long sRz,
    float* __restrict__ C, long ldcm, long ldcn, long sCz,
    int M, int N, int K, float scale, int act, int bT)
{
  __shared__ unsigned short lsA[BM_ * LDSA];
  __shared__ unsigned short lsB[BN_ * LDSB];

  const int z = blockIdx.z;
  A += (long)z * sAz;
  B += (long)z * sBz;
  C += (long)z * sCz;
  if (bias)  bias  += (long)z * sBiz;
  if (resid) resid += (long)z * sRz;

  const int tid    = threadIdx.x;
  const int wave   = tid >> 5;
  const int lane   = tid & 31;
  const int laneHi = lane >> 4;   // 0 / 1
  const int lane16 = lane & 15;

  const int m_blk = blockIdx.y * BM_;
  const int n_blk = blockIdx.x * BN_;
  const bool aFull = (m_blk + BM_ <= M);
  const bool nFull = (n_blk + BN_ <= N);

  v8f acc[2][4];
#pragma unroll
  for (int mi = 0; mi < 2; ++mi)
#pragma unroll
    for (int j = 0; j < 4; ++j)
#pragma unroll
      for (int e = 0; e < 8; ++e) acc[mi][j][e] = 0.0f;

  for (int k0 = 0; k0 < K; k0 += 32) {
    // ---- stage A tile (128 x 32) fp32 -> bf16 in LDS ----
    if (aFull) {
#pragma unroll
      for (int i = 0; i < 8; ++i) {              // 128 rows * 8 float4 cols
        int fid = tid + 128 * i;
        int r = fid >> 3, c4 = (fid & 7) * 4;
        const float4 v = *(const float4*)&A[(long)(m_blk + r) * lda + (k0 + c4)];
        unsigned int* dst = (unsigned int*)&lsA[r * LDSA + c4];
        dst[0] = pack_bf16x2(v.x, v.y);
        dst[1] = pack_bf16x2(v.z, v.w);
      }
    } else {
#pragma unroll
      for (int i = 0; i < 32; ++i) {
        int idx = tid + 128 * i;
        int r = idx >> 5, c = idx & 31;
        int gm = m_blk + r;
        float v = (gm < M) ? A[(long)gm * lda + (k0 + c)] : 0.0f;
        lsA[r * LDSA + c] = f32_to_bf16(v);
      }
    }
    // ---- stage B tile, stored [n][k] (32 x 64 logical) ----
    if (nFull) {
      if (bT) {
#pragma unroll
        for (int i = 0; i < 4; ++i) {            // 64 n-rows * 8 float4 cols
          int fid = tid + 128 * i;
          int nn = fid >> 3, c4 = (fid & 7) * 4;
          const float4 v = *(const float4*)&B[(long)(n_blk + nn) * ldb + (k0 + c4)];
          unsigned int* dst = (unsigned int*)&lsB[nn * LDSB + c4];
          dst[0] = pack_bf16x2(v.x, v.y);
          dst[1] = pack_bf16x2(v.z, v.w);
        }
      } else {
#pragma unroll
        for (int i = 0; i < 4; ++i) {            // 32 k-rows * 16 float4 cols
          int fid = tid + 128 * i;
          int kk = fid >> 4, n4 = (fid & 15) * 4;
          const float4 v = *(const float4*)&B[(long)(k0 + kk) * ldb + (n_blk + n4)];
          lsB[(n4 + 0) * LDSB + kk] = f32_to_bf16(v.x);
          lsB[(n4 + 1) * LDSB + kk] = f32_to_bf16(v.y);
          lsB[(n4 + 2) * LDSB + kk] = f32_to_bf16(v.z);
          lsB[(n4 + 3) * LDSB + kk] = f32_to_bf16(v.w);
        }
      }
    } else {
#pragma unroll
      for (int i = 0; i < 16; ++i) {
        int idx = tid + 128 * i;
        int kk = idx >> 6, nn = idx & 63;
        int gn = n_blk + nn;
        float v = 0.0f;
        if (gn < N) v = bT ? B[(long)gn * ldb + (k0 + kk)]
                           : B[(long)(k0 + kk) * ldb + gn];
        lsB[nn * LDSB + kk] = f32_to_bf16(v);
      }
    }
    // speculative prefetch of next K-tile (OOB-safe per ISA; dropped on fail)
    if (k0 + 32 < K) {
      __builtin_prefetch(&A[(long)(m_blk + tid) * lda + (k0 + 32)], 0, 0);
      if (bT) __builtin_prefetch(&B[(long)(n_blk + (tid & 63)) * ldb + (k0 + 32)], 0, 0);
      else    __builtin_prefetch(&B[(long)(k0 + 32 + (tid & 31)) * ldb + n_blk], 0, 0);
    }
    __syncthreads();

    // ---- fragments & 8 WMMAs ----
    union U { unsigned int u[8]; v16bf v; };
    const int arow0 = wave * 32 + lane16;
    U af[2];
#pragma unroll
    for (int mi = 0; mi < 2; ++mi)
#pragma unroll
      for (int r = 0; r < 8; ++r) {
        int kb = (r < 4 ? 2 * r : 8 + 2 * r) + (laneHi ? 8 : 0);
        af[mi].u[r] = *(const unsigned int*)&lsA[(arow0 + mi * 16) * LDSA + kb];
      }
#pragma unroll
    for (int j = 0; j < 4; ++j) {
      int ncol = j * 16 + lane16;
      U bfg;
#pragma unroll
      for (int r = 0; r < 8; ++r) {
        int kb = (r < 4 ? 2 * r : 8 + 2 * r) + (laneHi ? 8 : 0);
        bfg.u[r] = *(const unsigned int*)&lsB[ncol * LDSB + kb];
      }
      acc[0][j] = __builtin_amdgcn_wmma_f32_16x16x32_bf16(
          false, af[0].v, false, bfg.v, (short)0, acc[0][j], false, false);
      acc[1][j] = __builtin_amdgcn_wmma_f32_16x16x32_bf16(
          false, af[1].v, false, bfg.v, (short)0, acc[1][j], false, false);
    }
    __syncthreads();
  }

  // ---- epilogue: D layout: VGPR r -> M = r + 8*laneHi, N = lane16 ----
#pragma unroll
  for (int mi = 0; mi < 2; ++mi) {
    const int mrow0 = m_blk + wave * 32 + mi * 16 + laneHi * 8;
#pragma unroll
    for (int j = 0; j < 4; ++j) {
      int n = n_blk + j * 16 + lane16;
      if (n >= N) continue;
      float bb = bias ? bias[n] : 0.0f;
#pragma unroll
      for (int r = 0; r < 8; ++r) {
        int m = mrow0 + r;
        if (m >= M) continue;
        float t = acc[mi][j][r] + bb;
        if (act == 1)      t = t / (1.0f + __expf(-t));                    // SiLU
        else if (act == 2) t = 0.5f * t * (1.0f + erff(t * 0.70710678f));  // GELU
        t *= scale;
        if (resid) t += resid[(long)m * ldr + n];
        C[(long)m * ldcm + (long)n * ldcn] = t;
      }
    }
  }
}

// ---------------------------------------------------------------------------
// LayerNorm over last dim (block per row, in-place safe)
// ---------------------------------------------------------------------------
__global__ void ln_kernel(const float* __restrict__ x, long ldx,
                          const float* __restrict__ g, const float* __restrict__ b,
                          float* __restrict__ y, long ldy, int C)
{
  const int row = blockIdx.x;
  const float* xr = x + (long)row * ldx;
  float* yr = y + (long)row * ldy;
  __shared__ float red[256];
  float s = 0.f, sq = 0.f;
  for (int c = threadIdx.x; c < C; c += 256) { float v = xr[c]; s += v; sq += v * v; }
  red[threadIdx.x] = s; __syncthreads();
  for (int o = 128; o > 0; o >>= 1) { if (threadIdx.x < o) red[threadIdx.x] += red[threadIdx.x + o]; __syncthreads(); }
  float mean = red[0] / C; __syncthreads();
  red[threadIdx.x] = sq; __syncthreads();
  for (int o = 128; o > 0; o >>= 1) { if (threadIdx.x < o) red[threadIdx.x] += red[threadIdx.x + o]; __syncthreads(); }
  float var = red[0] / C - mean * mean;
  float rstd = rsqrtf(var + 1e-5f);
  for (int c = threadIdx.x; c < C; c += 256)
    yr[c] = (xr[c] - mean) * rstd * g[c] + b[c];
}

// ---------------------------------------------------------------------------
// Softmax with fused Transformer-XL relative shift:
//   score[t,s] = ac[h,t,s] + bd[h,t, (T-1) + s - t];  softmax over s, in-place.
// Grid: (T, H), 256 threads.
// ---------------------------------------------------------------------------
__global__ void softmax_relshift(float* __restrict__ ac, const float* __restrict__ bd)
{
  const int t = blockIdx.x, h = blockIdx.y;
  float* row = ac + ((long)h * T_ + t) * T_;
  const float* bdr = bd + ((long)h * T_ + t) * T2_ + (T_ - 1 - t);
  __shared__ float red[256];
  float mx = -1e30f;
  for (int s = threadIdx.x; s < T_; s += 256) mx = fmaxf(mx, row[s] + bdr[s]);
  red[threadIdx.x] = mx; __syncthreads();
  for (int o = 128; o > 0; o >>= 1) { if (threadIdx.x < o) red[threadIdx.x] = fmaxf(red[threadIdx.x], red[threadIdx.x + o]); __syncthreads(); }
  mx = red[0]; __syncthreads();
  float sum = 0.f;
  for (int s = threadIdx.x; s < T_; s += 256) {
    float e = __expf(row[s] + bdr[s] - mx);
    row[s] = e; sum += e;
  }
  red[threadIdx.x] = sum; __syncthreads();
  for (int o = 128; o > 0; o >>= 1) { if (threadIdx.x < o) red[threadIdx.x] += red[threadIdx.x + o]; __syncthreads(); }
  float inv = 1.0f / red[0];
  for (int s = threadIdx.x; s < T_; s += 256) row[s] *= inv;
}

// ---------------------------------------------------------------------------
// Depthwise conv along T, 'SAME', K=31. x,y: [T, C] row-major. addin: y += x.
// ---------------------------------------------------------------------------
__global__ void dwconv_kernel(const float* __restrict__ x, const float* __restrict__ w,
                              const float* __restrict__ b, float* __restrict__ y,
                              int C, int addin)
{
  long i = (long)blockIdx.x * 256 + threadIdx.x;
  if (i >= (long)T_ * C) return;
  int t = (int)(i / C), c = (int)(i % C);
  float acc = b[c];
#pragma unroll
  for (int j = 0; j < KS_; ++j) {
    int tt = t + j - (KS_ / 2);
    if (tt >= 0 && tt < T_) acc += w[c * KS_ + j] * x[(long)tt * C + c];
  }
  if (addin) acc += x[i];
  y[i] = acc;
}

// gated[t,c] = x2[t*2048 + c] * xg[t*1024 + c]   (c < 1024)
__global__ void gate_kernel(const float* __restrict__ x2, const float* __restrict__ xg,
                            float* __restrict__ out)
{
  long i = (long)blockIdx.x * 256 + threadIdx.x;
  if (i >= (long)T_ * CGH_) return;
  long t = i >> 10, c = i & 1023;
  out[i] = x2[t * (long)DCG_ + c] * xg[i];
}

// y[i] = x[i] + v[i % C]
__global__ void addrow_kernel(const float* __restrict__ x, const float* __restrict__ v,
                              float* __restrict__ y, long n, int C)
{
  long i = (long)blockIdx.x * 256 + threadIdx.x;
  if (i < n) y[i] = x[i] + v[i % C];
}

// pack [H, D, K] -> [D, H*K]
__global__ void pack_kernel(const float* __restrict__ in, float* __restrict__ out,
                            int Hh, int Dd, int Kk)
{
  long i = (long)blockIdx.x * 256 + threadIdx.x;
  long total = (long)Hh * Dd * Kk;
  if (i >= total) return;
  int h = (int)(i / ((long)Dd * Kk));
  int rem = (int)(i % ((long)Dd * Kk));
  int d = rem / Kk, k = rem % Kk;
  out[(long)d * (Hh * Kk) + h * Kk + k] = in[i];
}

// ---------------------------------------------------------------------------
// Host orchestration
// ---------------------------------------------------------------------------
static inline void launch_gemm(hipStream_t s,
    const float* A, long lda, long sAz,
    const float* B, long ldb, long sBz,
    const float* bias, long sBiz,
    const float* resid, long ldr, long sRz,
    float* C, long ldcm, long ldcn, long sCz,
    int M, int N, int K, float scale, int act, int bT, int Z)
{
  dim3 g((N + BN_ - 1) / BN_, (M + BM_ - 1) / BM_, Z);
  gemm_wmma<<<g, 128, 0, s>>>(A, lda, sAz, B, ldb, sBz, bias, sBiz,
                              resid, ldr, sRz, C, ldcm, ldcn, sCz,
                              M, N, K, scale, act, bT);
}

extern "C" void kernel_launch(void* const* d_in, const int* in_sizes, int n_in,
                              void* d_out, int out_size, void* d_ws, size_t ws_size,
                              hipStream_t stream)
{
  const float* X    = (const float*)d_in[0];
  const float* PE   = (const float*)d_in[1];   // [2T-1, D]
  const float* LNG  = (const float*)d_in[2];
  const float* LNB  = (const float*)d_in[3];
  const float* FFMW1= (const float*)d_in[4];
  const float* FFMB1= (const float*)d_in[5];
  const float* FFMW2= (const float*)d_in[6];
  const float* FFMB2= (const float*)d_in[7];
  const float* WQ   = (const float*)d_in[8];
  const float* BQ   = (const float*)d_in[9];
  const float* WK   = (const float*)d_in[10];
  const float* BK   = (const float*)d_in[11];
  const float* WV   = (const float*)d_in[12];
  const float* BV   = (const float*)d_in[13];
  const float* WPOS = (const float*)d_in[14];
  const float* PU   = (const float*)d_in[15];
  const float* PV   = (const float*)d_in[16];
  const float* WOUT = (const float*)d_in[17];
  const float* BOUT = (const float*)d_in[18];
  const float* CGW1 = (const float*)d_in[19];
  const float* CGB1 = (const float*)d_in[20];
  const float* CSG  = (const float*)d_in[21];
  const float* CSB  = (const float*)d_in[22];
  const float* CSCW = (const float*)d_in[23];
  const float* CSCB = (const float*)d_in[24];
  const float* CGW2 = (const float*)d_in[25];
  const float* CGB2 = (const float*)d_in[26];
  const float* FUSW = (const float*)d_in[27];
  const float* FUSB = (const float*)d_in[28];
  const float* MRGW = (const float*)d_in[29];
  const float* MRGB = (const float*)d_in[30];
  const float* FFW1 = (const float*)d_in[31];
  const float* FFB1 = (const float*)d_in[32];
  const float* FFW2 = (const float*)d_in[33];
  const float* FFB2 = (const float*)d_in[34];
  const float* AFG  = (const float*)d_in[35];
  const float* AFB  = (const float*)d_in[36];
  const float* DWK  = (const float*)d_in[37];
  const float* DBK  = (const float*)d_in[38];
  const float* DWV  = (const float*)d_in[39];
  const float* DBV  = (const float*)d_in[40];

  const long TD = (long)T_ * D_;
  float* ws = (float*)d_ws;
  size_t off = 0;
  auto alloc = [&](size_t n) { float* p = ws + off; off += n; return p; };
  float* x    = alloc(TD);
  float* xn   = alloc(TD);
  float* hid  = alloc((size_t)T_ * DFF_);
  float* qm   = alloc(TD);
  float* km   = alloc(TD);
  float* vm   = alloc(TD);
  float* qu   = alloc(TD);
  float* qv   = alloc(TD);
  float* pm   = alloc((size_t)T2_ * D_);
  float* pw   = alloc((size_t)D_ * D_);
  float* ac   = alloc((size_t)H_ * T_ * T_);
  float* bd   = alloc((size_t)H_ * T_ * T2_);
  float* ob   = alloc(TD);
  float* xc   = alloc((size_t)T_ * 2 * D_);
  float* cg   = alloc((size_t)T_ * DCG_);
  float* xgln = alloc((size_t)T_ * CGH_);
  float* xgcv = alloc((size_t)T_ * CGH_);
  float* gtd  = alloc((size_t)T_ * CGH_);
  float* xcf  = alloc((size_t)T_ * 2 * D_);
  (void)ws_size; (void)in_sizes; (void)n_in;

  hipMemcpyAsync(x, X, TD * sizeof(float), hipMemcpyDeviceToDevice, stream);

  auto ew_grid = [](long n) { return dim3((unsigned)((n + 255) / 256)); };

  for (int l = 0; l < L_; ++l) {
    const float* lg = LNG + ((long)l * 5) * D_;
    const float* lb = LNB + ((long)l * 5) * D_;

    // ---- macaron FF (half-step) ----
    ln_kernel<<<T_, 256, 0, stream>>>(x, D_, lg + 0 * D_, lb + 0 * D_, xn, D_, D_);
    launch_gemm(stream, xn, D_, 0, FFMW1 + (long)l * D_ * DFF_, DFF_, 0,
                FFMB1 + (long)l * DFF_, 0, nullptr, 0, 0,
                hid, DFF_, 1, 0, T_, DFF_, D_, 1.0f, 1 /*silu*/, 0, 1);
    launch_gemm(stream, hid, DFF_, 0, FFMW2 + (long)l * DFF_ * D_, D_, 0,
                FFMB2 + (long)l * D_, 0, x, D_, 0,
                x, D_, 1, 0, T_, D_, DFF_, 0.5f, 0, 0, 1);

    // ---- relative-position MHSA ----
    ln_kernel<<<T_, 256, 0, stream>>>(x, D_, lg + 1 * D_, lb + 1 * D_, xn, D_, D_);
    const long WSZ = (long)H_ * D_ * DK_;
    pack_kernel<<<ew_grid(WSZ), 256, 0, stream>>>(WQ + (long)l * WSZ, pw, H_, D_, DK_);
    launch_gemm(stream, xn, D_, 0, pw, D_, 0, BQ + (long)l * H_ * DK_, 0,
                nullptr, 0, 0, qm, D_, 1, 0, T_, D_, D_, 1.0f, 0, 0, 1);
    pack_kernel<<<ew_grid(WSZ), 256, 0, stream>>>(WK + (long)l * WSZ, pw, H_, D_, DK_);
    launch_gemm(stream, xn, D_, 0, pw, D_, 0, BK + (long)l * H_ * DK_, 0,
                nullptr, 0, 0, km, D_, 1, 0, T_, D_, D_, 1.0f, 0, 0, 1);
    pack_kernel<<<ew_grid(WSZ), 256, 0, stream>>>(WV + (long)l * WSZ, pw, H_, D_, DK_);
    launch_gemm(stream, xn, D_, 0, pw, D_, 0, BV + (long)l * H_ * DK_, 0,
                nullptr, 0, 0, vm, D_, 1, 0, T_, D_, D_, 1.0f, 0, 0, 1);
    pack_kernel<<<ew_grid(WSZ), 256, 0, stream>>>(WPOS + (long)l * WSZ, pw, H_, D_, DK_);
    launch_gemm(stream, PE, D_, 0, pw, D_, 0, nullptr, 0,
                nullptr, 0, 0, pm, D_, 1, 0, T2_, D_, D_, 1.0f, 0, 0, 1);

    addrow_kernel<<<ew_grid(TD), 256, 0, stream>>>(qm, PU + (long)l * H_ * DK_, qu, TD, D_);
    addrow_kernel<<<ew_grid(TD), 256, 0, stream>>>(qm, PV + (long)l * H_ * DK_, qv, TD, D_);

    // ac[h] = (q+u)_h @ k_h^T     [T,T]
    launch_gemm(stream, qu, D_, DK_, km, D_, DK_, nullptr, 0, nullptr, 0, 0,
                ac, T_, 1, (long)T_ * T_, T_, T_, DK_, 1.0f, 0, 1 /*NT*/, H_);
    // bd[h] = (q+v)_h @ p_h^T     [T,2T-1]
    launch_gemm(stream, qv, D_, DK_, pm, D_, DK_, nullptr, 0, nullptr, 0, 0,
                bd, T2_, 1, (long)T_ * T2_, T_, T2_, DK_, 1.0f, 0, 1 /*NT*/, H_);

    softmax_relshift<<<dim3(T_, H_), 256, 0, stream>>>(ac, bd);

    // o_h = attn_h @ v_h -> ob as [T, H*DK]
    launch_gemm(stream, ac, T_, (long)T_ * T_, vm, D_, DK_, nullptr, 0, nullptr, 0, 0,
                ob, D_, 1, DK_, T_, DK_, T_, 1.0f, 0, 0, H_);
    // x1 = o @ Wout + bout  -> xc[:, 0:D]
    launch_gemm(stream, ob, D_, 0, WOUT + (long)l * D_ * D_, D_, 0,
                BOUT + (long)l * D_, 0, nullptr, 0, 0,
                xc, 2 * D_, 1, 0, T_, D_, D_, 1.0f, 0, 0, 1);

    // ---- cgMLP branch ----
    ln_kernel<<<T_, 256, 0, stream>>>(x, D_, lg + 2 * D_, lb + 2 * D_, xn, D_, D_);
    launch_gemm(stream, xn, D_, 0, CGW1 + (long)l * D_ * DCG_, DCG_, 0,
                CGB1 + (long)l * DCG_, 0, nullptr, 0, 0,
                cg, DCG_, 1, 0, T_, DCG_, D_, 1.0f, 2 /*gelu*/, 0, 1);
    // LN on gate half (columns CGH..2CGH of cg)
    ln_kernel<<<T_, 256, 0, stream>>>(cg + CGH_, DCG_, CSG + (long)l * CGH_,
                                      CSB + (long)l * CGH_, xgln, CGH_, CGH_);
    dwconv_kernel<<<ew_grid((long)T_ * CGH_), 256, 0, stream>>>(
        xgln, CSCW + (long)l * CGH_ * KS_, CSCB + (long)l * CGH_, xgcv, CGH_, 0);
    gate_kernel<<<ew_grid((long)T_ * CGH_), 256, 0, stream>>>(cg, xgcv, gtd);
    // x2 = gated @ cg_w2 + cg_b2 -> xc[:, D:2D]
    launch_gemm(stream, gtd, CGH_, 0, CGW2 + (long)l * CGH_ * D_, D_, 0,
                CGB2 + (long)l * D_, 0, nullptr, 0, 0,
                xc + D_, 2 * D_, 1, 0, T_, D_, CGH_, 1.0f, 0, 0, 1);

    // ---- branch fusion: xcf = xc + dwconv(xc) ; x += xcf @ merge + b ----
    dwconv_kernel<<<ew_grid((long)T_ * 2 * D_), 256, 0, stream>>>(
        xc, FUSW + (long)l * 2 * D_ * KS_, FUSB + (long)l * 2 * D_, xcf, 2 * D_, 1);
    launch_gemm(stream, xcf, 2 * D_, 0, MRGW + (long)l * 2 * D_ * D_, D_, 0,
                MRGB + (long)l * D_, 0, x, D_, 0,
                x, D_, 1, 0, T_, D_, 2 * D_, 1.0f, 0, 0, 1);

    // ---- final FF (half-step) + layer LN ----
    ln_kernel<<<T_, 256, 0, stream>>>(x, D_, lg + 3 * D_, lb + 3 * D_, xn, D_, D_);
    launch_gemm(stream, xn, D_, 0, FFW1 + (long)l * D_ * DFF_, DFF_, 0,
                FFB1 + (long)l * DFF_, 0, nullptr, 0, 0,
                hid, DFF_, 1, 0, T_, DFF_, D_, 1.0f, 1, 0, 1);
    launch_gemm(stream, hid, DFF_, 0, FFW2 + (long)l * DFF_ * D_, D_, 0,
                FFB2 + (long)l * D_, 0, x, D_, 0,
                x, D_, 1, 0, T_, D_, DFF_, 0.5f, 0, 0, 1);
    ln_kernel<<<T_, 256, 0, stream>>>(x, D_, lg + 4 * D_, lb + 4 * D_, x, D_, D_);
  }

  // ---- after-norm + decoder K/V projections ----
  ln_kernel<<<T_, 256, 0, stream>>>(x, D_, AFG, AFB, xn, D_, D_);  // xn = enc
  float* keys = (float*)d_out;                                     // [NDE,H,DK,T]
  float* vals = keys + (long)NDE_ * H_ * DK_ * T_;                 // [NDE,H,T,DK]
  const long WSZ = (long)H_ * D_ * DK_;
  for (int n = 0; n < NDE_; ++n) {
    pack_kernel<<<dim3((unsigned)((WSZ + 255) / 256)), 256, 0, stream>>>(
        DWK + (long)n * WSZ, pw, H_, D_, DK_);
    // keys: transposed store (m=t stride 1, n=k stride T)
    launch_gemm(stream, xn, D_, 0, pw, D_, DK_, DBK + (long)n * H_ * DK_, DK_,
                nullptr, 0, 0, keys + (long)n * H_ * DK_ * T_,
                1, T_, (long)DK_ * T_, T_, DK_, D_, 1.0f, 0, 0, H_);
    pack_kernel<<<dim3((unsigned)((WSZ + 255) / 256)), 256, 0, stream>>>(
        DWV + (long)n * WSZ, pw, H_, D_, DK_);
    launch_gemm(stream, xn, D_, 0, pw, D_, DK_, DBV + (long)n * H_ * DK_, DK_,
                nullptr, 0, 0, vals + (long)n * H_ * T_ * DK_,
                DK_, 1, (long)T_ * DK_, T_, DK_, D_, 1.0f, 0, 0, H_);
  }
  (void)out_size;
}
